// StackRNN_38122129719948
// MI455X (gfx1250) — matrix operand order repaired and checked by reading, not compiled, not run on hardware
//
#include <hip/hip_runtime.h>

#define B_DIM 8192
#define IDIM  512
#define HDIM  1024
#define ODIM  512
#define S_DIM 64
#define DEPTH 64
#define KC    128            // K-chunk staged in LDS per TDM transfer

typedef __attribute__((ext_vector_type(16))) __bf16          v16bf;
typedef __attribute__((ext_vector_type(16))) unsigned short  v16u;
typedef __attribute__((ext_vector_type(8)))  unsigned short  v8u;
typedef __attribute__((ext_vector_type(4)))  unsigned short  u16x4;
typedef __attribute__((ext_vector_type(8)))  float           v8f;
typedef __attribute__((ext_vector_type(4)))  float           f32x4;
typedef __attribute__((ext_vector_type(4)))  unsigned int    u32x4;
typedef __attribute__((ext_vector_type(8)))  int             i32x8;
typedef __attribute__((ext_vector_type(4)))  int             i32x4;

union Frag { v16u u; v16bf bf; };

__device__ __forceinline__ unsigned short f2bf(float f) {
    union { float f; unsigned int u; } c; c.f = f;
    unsigned int u = c.u;
    u += 0x7FFFu + ((u >> 16) & 1u);   // round-to-nearest-even
    return (unsigned short)(u >> 16);
}

// ---------------------------------------------------------------------------
// TDM: DMA a [64 rows x chunk cols] bf16 tile of W[N,K] (rows n0..n0+63,
// cols kc..kc+chunk) into LDS at lds_dst, with 16B padding per row so the
// LDS row stride is (chunk+8) elements (breaks ds bank aliasing).
// Descriptor packing per CDNA5 ISA.
// ---------------------------------------------------------------------------
__device__ __forceinline__ void tdm_load_b(const unsigned short* W, int K, int n0,
                                           int kc, int chunk,
                                           const unsigned short* lds_dst) {
    unsigned long long gaddr = (unsigned long long)(size_t)(W + (size_t)n0 * K + kc);
    unsigned int lds_addr = (unsigned int)(size_t)(const void*)lds_dst; // LDS aperture: addr[31:0]
    int pad_interval = __builtin_ctz((unsigned)chunk) - 2;  // (1<<iv)*8B == chunk*2B
    u32x4 g0;
    g0[0] = 1u;                                             // count=1, user mode
    g0[1] = lds_addr;                                       // lds_addr
    g0[2] = (unsigned int)gaddr;                            // global_addr[31:0]
    g0[3] = (unsigned int)((gaddr >> 32) & 0x1FFFFFFull)    // global_addr[56:32]
            | 0x80000000u;                                  // type=2 ("image")
    i32x8 g1;
    g1[0] = (1 << 16)                                       // data_size = 2 bytes
          | (1 << 20)                                       // pad_enable
          | (pad_interval << 22)                            // pad every row
          | (3 << 25);                                      // pad_amount = 4 dwords (16B)
    g1[1] = (chunk & 0xFFFF) << 16;                         // tensor_dim0[15:0]
    g1[2] = ((chunk >> 16) & 0xFFFF) | (64 << 16);          // tensor_dim0 hi | tensor_dim1=64
    g1[3] = (chunk & 0xFFFF) << 16;                         // tensor_dim1 hi=0 | tile_dim0=chunk
    g1[4] = 64;                                             // tile_dim1=64, tile_dim2=0
    g1[5] = K;                                              // tensor_dim0_stride[31:0] (elements)
    g1[6] = 0;                                              // stride hi | dim1_stride lo
    g1[7] = 0;
    i32x4 z4 = {0, 0, 0, 0};
#if defined(__clang_major__) && (__clang_major__ >= 23)
    i32x8 z8 = {0, 0, 0, 0, 0, 0, 0, 0};
    __builtin_amdgcn_tensor_load_to_lds(g0, g1, z4, z4, z8, 0);
#else
    __builtin_amdgcn_tensor_load_to_lds(g0, g1, z4, z4, 0);
#endif
}

// ---------------------------------------------------------------------------
// f32 -> bf16 conversion pass.  row_len is a power of two (shift given).
// ---------------------------------------------------------------------------
__global__ __launch_bounds__(256)
void cvt_bf16_kernel(const float* __restrict__ src, unsigned short* __restrict__ dst,
                     long total, int src_stride, int len_shift) {
    long e = ((long)blockIdx.x * 256 + threadIdx.x) * 4;
    if (e >= total) return;
    long row = e >> len_shift;
    int  col = (int)(e & ((1 << len_shift) - 1));
    f32x4 v = *(const f32x4*)(src + row * src_stride + col);
    u16x4 o;
#pragma unroll
    for (int j = 0; j < 4; ++j) o[j] = f2bf(v[j]);
    *(u16x4*)(dst + e) = o;
}

// ---------------------------------------------------------------------------
// WMMA GEMM with TDM-staged B panel.
// C[M,N] = act( sum_seg A_seg(bf16) @ W_seg(bf16)^T + sum bias )
// Block: 256 threads = 8 wave32 stacked in M; wave tile 32(M)x64(N);
// block tile 256x64.  B panel [64 x KC] double-buffered in LDS via TDM.
// ---------------------------------------------------------------------------
struct Seg { const unsigned short* A; const unsigned short* W; const float* bias; int K; int lda; };
struct GemmArgs { Seg s[4]; int nseg; };

__global__ __launch_bounds__(256)
void wmma_gemm_kernel(GemmArgs args, float* __restrict__ C,
                      unsigned short* __restrict__ Cbf, int N, int act) {
    __shared__ unsigned short bsm[2][64 * (KC + 8)];

    const int lane = threadIdx.x & 31;
    const int wid  = threadIdx.x >> 5;
    const int m0   = blockIdx.y * 256 + wid * 32;
    const int n0   = blockIdx.x * 64;
    const int r    = lane & 15;
    const int hf   = lane >> 4;

    v8f c[2][4] = {};

    for (int s = 0; s < args.nseg; ++s) {
        const unsigned short* A = args.s[s].A;
        const unsigned short* W = args.s[s].W;
        const int K   = args.s[s].K;
        const int lda = args.s[s].lda;
        // A layout (16-bit A 16x32): lane r<16 -> M=r, K runs
        // contiguous {k+hf*16 .. +15} (matches B).
        const unsigned short* ar0 = A + (size_t)(m0 + r) * lda + hf * 16;
        const unsigned short* ar1 = ar0 + (size_t)16 * lda;

        const int nchunks = (K + KC - 1) / KC;
        if (wid == 0)
            tdm_load_b(W, K, n0, 0, (K < KC ? K : KC), &bsm[0][0]);

        for (int ci = 0; ci < nchunks; ++ci) {
            const int kc    = ci * KC;
            const int chunk = (K - kc < KC) ? (K - kc) : KC;
            if (wid == 0) __builtin_amdgcn_s_wait_tensorcnt(0); // chunk ci landed
            __syncthreads();                                    // visible to all waves
            if (wid == 0 && ci + 1 < nchunks) {                 // prefetch next chunk (other buffer)
                const int kc2 = kc + KC;
                const int ch2 = (K - kc2 < KC) ? (K - kc2) : KC;
                tdm_load_b(W, K, n0, kc2, ch2, &bsm[(ci + 1) & 1][0]);
            }
            const int ldb = chunk + 8;                          // padded LDS row stride (elements)
            // B layout (KxN from W[N,K]): lane holds contiguous 16-K run of row n.
            const unsigned short* bp = &bsm[ci & 1][0] + r * ldb + hf * 16;

            for (int k0 = 0; k0 < chunk; k0 += 32) {
                Frag a0, a1, b0, b1, b2, b3;
                const unsigned short* a0p = ar0 + kc + k0;
                const unsigned short* a1p = ar1 + kc + k0;
                a0.u = __builtin_shufflevector(*(const v8u*)(a0p), *(const v8u*)(a0p + 8),
                                               0,1,2,3,4,5,6,7,8,9,10,11,12,13,14,15);
                a1.u = __builtin_shufflevector(*(const v8u*)(a1p), *(const v8u*)(a1p + 8),
                                               0,1,2,3,4,5,6,7,8,9,10,11,12,13,14,15);
                const unsigned short* b0p = bp + k0;
                const unsigned short* b1p = b0p + 16 * ldb;
                const unsigned short* b2p = b0p + 32 * ldb;
                const unsigned short* b3p = b0p + 48 * ldb;
                b0.u = __builtin_shufflevector(*(const v8u*)(b0p), *(const v8u*)(b0p + 8),
                                               0,1,2,3,4,5,6,7,8,9,10,11,12,13,14,15);
                b1.u = __builtin_shufflevector(*(const v8u*)(b1p), *(const v8u*)(b1p + 8),
                                               0,1,2,3,4,5,6,7,8,9,10,11,12,13,14,15);
                b2.u = __builtin_shufflevector(*(const v8u*)(b2p), *(const v8u*)(b2p + 8),
                                               0,1,2,3,4,5,6,7,8,9,10,11,12,13,14,15);
                b3.u = __builtin_shufflevector(*(const v8u*)(b3p), *(const v8u*)(b3p + 8),
                                               0,1,2,3,4,5,6,7,8,9,10,11,12,13,14,15);
                if (kc + k0 + 32 < K) {            // gfx1250 global_prefetch_b8 for A stream
                    __builtin_prefetch(a0p + 32, 0, 1);
                    __builtin_prefetch(a1p + 32, 0, 1);
                }
                c[0][0] = __builtin_amdgcn_wmma_f32_16x16x32_bf16(false, a0.bf, false, b0.bf, (short)0, c[0][0], false, false);
                c[0][1] = __builtin_amdgcn_wmma_f32_16x16x32_bf16(false, a0.bf, false, b1.bf, (short)0, c[0][1], false, false);
                c[0][2] = __builtin_amdgcn_wmma_f32_16x16x32_bf16(false, a0.bf, false, b2.bf, (short)0, c[0][2], false, false);
                c[0][3] = __builtin_amdgcn_wmma_f32_16x16x32_bf16(false, a0.bf, false, b3.bf, (short)0, c[0][3], false, false);
                c[1][0] = __builtin_amdgcn_wmma_f32_16x16x32_bf16(false, a1.bf, false, b0.bf, (short)0, c[1][0], false, false);
                c[1][1] = __builtin_amdgcn_wmma_f32_16x16x32_bf16(false, a1.bf, false, b1.bf, (short)0, c[1][1], false, false);
                c[1][2] = __builtin_amdgcn_wmma_f32_16x16x32_bf16(false, a1.bf, false, b2.bf, (short)0, c[1][2], false, false);
                c[1][3] = __builtin_amdgcn_wmma_f32_16x16x32_bf16(false, a1.bf, false, b3.bf, (short)0, c[1][3], false, false);
            }
        }
        __syncthreads();   // all reads done before next segment reuses buffers
    }

    // Epilogue. C/D f32 layout: lane column = n0 + j*16 + r; rows = m0 + i*16 + hf*8 + v.
    const int ncol = n0 + r;
    float bsum[4] = {0.f, 0.f, 0.f, 0.f};
    for (int s = 0; s < args.nseg; ++s) {
        const float* bias = args.s[s].bias;
        if (bias) {
#pragma unroll
            for (int j = 0; j < 4; ++j) bsum[j] += bias[ncol + j * 16];
        }
    }
    const int mbase = m0 + hf * 8;
#pragma unroll
    for (int i = 0; i < 2; ++i) {
#pragma unroll
        for (int j = 0; j < 4; ++j) {
#pragma unroll
            for (int v = 0; v < 8; ++v) {
                float val = c[i][j][v] + bsum[j];
                if (act == 1) val = 1.0f / (1.0f + __expf(-val));
                size_t idx = (size_t)(mbase + i * 16 + v) * N + ncol + j * 16;
                C[idx] = val;
                if (Cbf) Cbf[idx] = f2bf(val);
            }
        }
    }
}

// softmax over the 3 gate groups of raw[B,192] -> act0/1/2 [B*64]
__global__ __launch_bounds__(256)
void softmax3_kernel(const float* __restrict__ raw,
                     float* __restrict__ a0, float* __restrict__ a1, float* __restrict__ a2) {
    int i = blockIdx.x * blockDim.x + threadIdx.x;
    if (i >= B_DIM * S_DIM) return;
    int b = i >> 6, s = i & 63;
    const float* rr = raw + (size_t)b * 192;
    float x0 = rr[s], x1 = rr[64 + s], x2 = rr[128 + s];
    float mx = fmaxf(x0, fmaxf(x1, x2));
    float e0 = __expf(x0 - mx), e1 = __expf(x1 - mx), e2 = __expf(x2 - mx);
    float inv = 1.0f / (e0 + e1 + e2);
    a0[i] = e0 * inv; a1[i] = e1 * inv; a2[i] = e2 * inv;
}

// new_stack = push*a0 + pop*a1 + stack*a2 ; float4 over the contiguous s dim.
__global__ __launch_bounds__(256)
void stack_update_kernel(const float* __restrict__ stk,
                         const float* __restrict__ a0, const float* __restrict__ a1,
                         const float* __restrict__ a2, const float* __restrict__ top,
                         float* __restrict__ ns) {
    int i = blockIdx.x * blockDim.x + threadIdx.x;     // (b, d, s/4)
    if (i >= B_DIM * DEPTH * (S_DIM / 4)) return;
    int s4 = i & 15;
    int d  = (i >> 4) & 63;
    int b  = i >> 10;
    size_t base = ((size_t)b * DEPTH + d) * S_DIM + s4 * 4;
    f32x4 cur = *(const f32x4*)(stk + base);
    f32x4 push = (d == 0) ? *(const f32x4*)(top + (size_t)b * S_DIM + s4 * 4)
                          : *(const f32x4*)(stk + base - S_DIM);
    f32x4 pop = (d == DEPTH - 1) ? (f32x4){-1.f, -1.f, -1.f, -1.f}
                                 : *(const f32x4*)(stk + base + S_DIM);
    size_t ai = (size_t)b * S_DIM + s4 * 4;
    f32x4 p = *(const f32x4*)(a0 + ai);
    f32x4 q = *(const f32x4*)(a1 + ai);
    f32x4 t = *(const f32x4*)(a2 + ai);
    f32x4 res;
#pragma unroll
    for (int cidx = 0; cidx < 4; ++cidx)
        res[cidx] = push[cidx] * p[cidx] + pop[cidx] * q[cidx] + cur[cidx] * t[cidx];
    *(f32x4*)(ns + base) = res;
}

// ---------------------------------------------------------------------------

static inline void launch_cvt(const float* src, unsigned short* dst, long rows,
                              int src_stride, int row_len, int shift, hipStream_t stream) {
    long total = rows * row_len;
    int blocks = (int)((total / 4 + 255) / 256);
    cvt_bf16_kernel<<<blocks, 256, 0, stream>>>(src, dst, total, src_stride, shift);
}

extern "C" void kernel_launch(void* const* d_in, const int* in_sizes, int n_in,
                              void* d_out, int out_size, void* d_ws, size_t ws_size,
                              hipStream_t stream) {
    const float* x      = (const float*)d_in[0];
    const float* hidden = (const float*)d_in[1];
    const float* stk    = (const float*)d_in[2];
    const float* Wi = (const float*)d_in[3];  const float* bi = (const float*)d_in[4];
    const float* Wh = (const float*)d_in[5];  const float* bh = (const float*)d_in[6];
    const float* Wo = (const float*)d_in[7];  const float* bo = (const float*)d_in[8];
    const float* Wa = (const float*)d_in[9];  const float* ba = (const float*)d_in[10];
    const float* Ws = (const float*)d_in[11]; const float* bs = (const float*)d_in[12];
    const float* W0 = (const float*)d_in[13]; const float* b0 = (const float*)d_in[14];
    const float* W1 = (const float*)d_in[15]; const float* b1 = (const float*)d_in[16];

    // d_out = [out B*512 | h B*1024 | new_stack B*64*64]
    float* out = (float*)d_out;
    float* h   = out + (size_t)B_DIM * ODIM;
    float* nst = h   + (size_t)B_DIM * HDIM;

    // ---- workspace layout -------------------------------------------------
    float* rawA = (float*)d_ws;                         // [B,192] f32
    float* top  = rawA + (size_t)B_DIM * 192;           // [B,64]  f32
    float* a0   = top  + (size_t)B_DIM * S_DIM;
    float* a1   = a0   + (size_t)B_DIM * S_DIM;
    float* a2   = a1   + (size_t)B_DIM * S_DIM;
    unsigned short* xbf   = (unsigned short*)(a2 + (size_t)B_DIM * S_DIM);
    unsigned short* hidbf = xbf   + (size_t)B_DIM * IDIM;
    unsigned short* st0bf = hidbf + (size_t)B_DIM * HDIM;
    unsigned short* st1bf = st0bf + (size_t)B_DIM * S_DIM;
    unsigned short* hbf   = st1bf + (size_t)B_DIM * S_DIM;
    unsigned short* Wibf  = hbf   + (size_t)B_DIM * HDIM;
    unsigned short* Whbf  = Wibf  + (size_t)HDIM * IDIM;
    unsigned short* W0bf  = Whbf  + (size_t)HDIM * HDIM;
    unsigned short* W1bf  = W0bf  + (size_t)HDIM * S_DIM;
    unsigned short* Wabf  = W1bf  + (size_t)HDIM * S_DIM;
    unsigned short* Wsbf  = Wabf  + (size_t)(3 * S_DIM) * HDIM;
    unsigned short* Wobf  = Wsbf  + (size_t)S_DIM * HDIM;

    // ---- pass 0: one-time f32 -> bf16 conversions -------------------------
    launch_cvt(x,        xbf,   B_DIM,     IDIM,          IDIM,  9,  stream);
    launch_cvt(hidden,   hidbf, B_DIM,     HDIM,          HDIM,  10, stream);
    launch_cvt(stk,      st0bf, B_DIM,     DEPTH * S_DIM, S_DIM, 6,  stream);  // stack[:,0,:]
    launch_cvt(stk + 64, st1bf, B_DIM,     DEPTH * S_DIM, S_DIM, 6,  stream);  // stack[:,1,:]
    launch_cvt(Wi,       Wibf,  HDIM,      IDIM,          IDIM,  9,  stream);
    launch_cvt(Wh,       Whbf,  HDIM,      HDIM,          HDIM,  10, stream);
    launch_cvt(W0,       W0bf,  HDIM,      S_DIM,         S_DIM, 6,  stream);
    launch_cvt(W1,       W1bf,  HDIM,      S_DIM,         S_DIM, 6,  stream);
    launch_cvt(Wa,       Wabf,  3 * S_DIM, HDIM,          HDIM,  10, stream);
    launch_cvt(Ws,       Wsbf,  S_DIM,     HDIM,          HDIM,  10, stream);
    launch_cvt(Wo,       Wobf,  ODIM,      HDIM,          HDIM,  10, stream);

    dim3 blk(256);

    // 1) h = sigmoid(x@Wi^T + hidden@Wh^T + st0@W0^T + st1@W1^T + biases)
    GemmArgs g1 = {};
    g1.s[0] = { xbf,   Wibf, bi, IDIM,  IDIM  };
    g1.s[1] = { hidbf, Whbf, bh, HDIM,  HDIM  };
    g1.s[2] = { st0bf, W0bf, b0, S_DIM, S_DIM };
    g1.s[3] = { st1bf, W1bf, b1, S_DIM, S_DIM };
    g1.nseg = 4;
    wmma_gemm_kernel<<<dim3(HDIM / 64, B_DIM / 256), blk, 0, stream>>>(g1, h, hbf, HDIM, 1);

    // 2) rawA = h@Wa^T + ba   [B,192]
    GemmArgs g2 = {}; g2.s[0] = { hbf, Wabf, ba, HDIM, HDIM }; g2.nseg = 1;
    wmma_gemm_kernel<<<dim3(192 / 64, B_DIM / 256), blk, 0, stream>>>(g2, rawA, nullptr, 192, 0);

    // 3) top = sigmoid(h@Ws^T + bs)   [B,64]
    GemmArgs g3 = {}; g3.s[0] = { hbf, Wsbf, bs, HDIM, HDIM }; g3.nseg = 1;
    wmma_gemm_kernel<<<dim3(S_DIM / 64, B_DIM / 256), blk, 0, stream>>>(g3, top, nullptr, S_DIM, 1);

    // 4) act = softmax(rawA over 3 groups)
    {
        int n = B_DIM * S_DIM;
        softmax3_kernel<<<(n + 255) / 256, blk, 0, stream>>>(rawA, a0, a1, a2);
    }

    // 5) new_stack update (main HBM stream: 256 MB)
    {
        int n = B_DIM * DEPTH * (S_DIM / 4);
        stack_update_kernel<<<(n + 255) / 256, blk, 0, stream>>>(stk, a0, a1, a2, top, nst);
    }

    // 6) out = h@Wo^T + bo
    GemmArgs g6 = {}; g6.s[0] = { hbf, Wobf, bo, HDIM, HDIM }; g6.nseg = 1;
    wmma_gemm_kernel<<<dim3(ODIM / 64, B_DIM / 256), blk, 0, stream>>>(g6, out, ODIM == 0 ? nullptr : nullptr, ODIM, 0);
}
